// CocycleAttention_32796370272656
// MI455X (gfx1250) — matrix operand order repaired and checked by reading, not compiled, not run on hardware
//
#include <hip/hip_runtime.h>

// ---------------------------------------------------------------------------
// CocycleAttention on MI455X (gfx1250, wave32, WMMA), v2.
//
// Pipeline (fp32 I/O, bf16 WMMA operands, fp32 accumulate):
//   1) proj_gemm    : Qp/Kp/Vp = X @ W^T + b            (v_wmma_f32_16x16x32_bf16)
//   2) logsoftmax_q : logpQ (bf16) from Qp
//   3) logsoftmax_k : pK (bf16) + neg_ent (f32) from Kp
//   4) scores_gemm  : attn_raw[i,j] = logpQ_i . pK_j - neg_ent[j] -> d_out attn
//   5) attn_softmax : softmax over j, in place in d_out
//   6) out_gemm     : out = attn @ Vp                    -> d_out out region
//
// v2: 128x128 block tile, 256 threads (8 wave32s), wave tile 32x64 (8 WMMA /
// K-step / wave), double-buffered LDS staging (40 KB), global prefetch one
// slab ahead.
// ---------------------------------------------------------------------------

typedef __attribute__((ext_vector_type(16))) __bf16 v16bf;
typedef __attribute__((ext_vector_type(8)))  float  v8f;

#define BATCH  4
#define NSEQ   2048
#define DMODEL 1024
#define LDSS   40            // LDS row stride (bf16 elems); 40 breaks bank conflicts
#define BUFE   (128 * LDSS)  // one LDS slab (128 rows x 32 k)

__device__ __forceinline__ v8f wmma_bf16(v16bf a, v16bf b, v8f c) {
  // (neg_a, A, neg_b, B, c_mod, C, reuse_a, reuse_b)
  return __builtin_amdgcn_wmma_f32_16x16x32_bf16(false, a, false, b, (short)0, c, false, false);
}

// A fragment: 16x32 bf16. lane m (0..15) = row m; lanes 16..31 same rows, K +8.
__device__ __forceinline__ v16bf load_a_frag(const __bf16* tile, int ar, int lane) {
  const int m  = ar + (lane & 15);
  const int kh = (lane >> 4) * 8;
  const __bf16* p = tile + m * LDSS;
  v16bf a;
#pragma unroll
  for (int e = 0; e < 8; ++e) a[e] = p[kh + e];
#pragma unroll
  for (int e = 0; e < 8; ++e) a[8 + e] = p[16 + kh + e];
  return a;
}

// B fragment: 32x16 bf16 (KxN). lane n (0..15) = column n; lanes 16..31 same cols, K +8.
__device__ __forceinline__ v16bf load_b_frag(const __bf16* tile, int bc, int lane) {
  const int n  = bc + (lane & 15);
  const int kh = (lane >> 4) * 8;
  const __bf16* p = tile + n * LDSS;
  v16bf b;
#pragma unroll
  for (int e = 0; e < 8; ++e) b[e] = p[kh + e];
#pragma unroll
  for (int e = 0; e < 8; ++e) b[8 + e] = p[16 + kh + e];
  return b;
}

// ---------------------------------------------------------------------------
// Kernel 1: P[m,n] = sum_k X[m,k] * W[n,k] + bias[n]    (M=8192, N=K=1024)
// ---------------------------------------------------------------------------
__global__ void __launch_bounds__(256)
proj_gemm(const float* __restrict__ X, const float* __restrict__ W,
          const float* __restrict__ bias, float* __restrict__ P) {
  __shared__ __align__(16) __bf16 As[2 * BUFE];
  __shared__ __align__(16) __bf16 Bs[2 * BUFE];

  const int tileN = blockIdx.x * 128;
  const int tileM = blockIdx.y * 128;
  const int t    = threadIdx.x;
  const int lane = t & 31;
  const int wv   = t >> 5;
  const int wr   = (wv >> 1) * 32;   // wave row offset (4 row-waves)
  const int wc   = (wv & 1) * 64;    // wave col offset (2 col-waves)

  auto stage = [&](int buf, int kb) {
    __bf16* A = As + buf * BUFE;
    __bf16* B = Bs + buf * BUFE;
#pragma unroll
    for (int it = 0; it < 4; ++it) {        // 128 rows x 8 float4, 4 per thread
      const int idx = t + it * 256;
      const int row = idx >> 3;
      const int c4  = (idx & 7) * 4;
      float4 va = *(const float4*)(X + (size_t)(tileM + row) * DMODEL + kb + c4);
      __bf16* da = A + row * LDSS + c4;
      da[0] = (__bf16)va.x; da[1] = (__bf16)va.y; da[2] = (__bf16)va.z; da[3] = (__bf16)va.w;
      float4 vb = *(const float4*)(W + (size_t)(tileN + row) * DMODEL + kb + c4);
      __bf16* db = B + row * LDSS + c4;
      db[0] = (__bf16)vb.x; db[1] = (__bf16)vb.y; db[2] = (__bf16)vb.z; db[3] = (__bf16)vb.w;
    }
  };

  v8f acc[2][4] = {};
  stage(0, 0);
  int p = 0;
  for (int kb = 0; kb < DMODEL; kb += 32) {
    __syncthreads();
    if (kb + 32 < DMODEL) stage(p ^ 1, kb + 32);          // overlap next slab
    if (kb + 64 < DMODEL) {                               // global_prefetch_b8
      __builtin_prefetch(X + (size_t)(tileM + (t & 127)) * DMODEL + kb + 64, 0, 3);
      __builtin_prefetch(W + (size_t)(tileN + (t & 127)) * DMODEL + kb + 64, 0, 3);
    }
    const __bf16* A = As + p * BUFE;
    const __bf16* B = Bs + p * BUFE;
    v16bf af[2], bfrag[4];
    af[0] = load_a_frag(A, wr,      lane);
    af[1] = load_a_frag(A, wr + 16, lane);
#pragma unroll
    for (int n = 0; n < 4; ++n) bfrag[n] = load_b_frag(B, wc + n * 16, lane);
#pragma unroll
    for (int m = 0; m < 2; ++m)
#pragma unroll
      for (int n = 0; n < 4; ++n) acc[m][n] = wmma_bf16(af[m], bfrag[n], acc[m][n]);
    p ^= 1;
  }

  const int kh   = lane >> 4;
  const int nloc = lane & 15;
#pragma unroll
  for (int m = 0; m < 2; ++m)
#pragma unroll
    for (int n = 0; n < 4; ++n) {
      const int gcol = tileN + wc + n * 16 + nloc;
      const float bv = bias[gcol];
#pragma unroll
      for (int e = 0; e < 8; ++e) {
        const int grow = tileM + wr + m * 16 + kh * 8 + e;
        P[(size_t)grow * DMODEL + gcol] = acc[m][n][e] + bv;
      }
    }
}

// ---------------------------------------------------------------------------
// Block reductions (256 threads)
// ---------------------------------------------------------------------------
__device__ __forceinline__ float block_max(float v, float* sm) {
  const int t = threadIdx.x;
  sm[t] = v; __syncthreads();
  for (int s = 128; s > 0; s >>= 1) { if (t < s) sm[t] = fmaxf(sm[t], sm[t + s]); __syncthreads(); }
  float r = sm[0]; __syncthreads(); return r;
}
__device__ __forceinline__ float block_sum(float v, float* sm) {
  const int t = threadIdx.x;
  sm[t] = v; __syncthreads();
  for (int s = 128; s > 0; s >>= 1) { if (t < s) sm[t] += sm[t + s]; __syncthreads(); }
  float r = sm[0]; __syncthreads(); return r;
}

// ---------------------------------------------------------------------------
// Kernel 2: logpQ = log_softmax(Qp) rows of length 1024, stored bf16.
// ---------------------------------------------------------------------------
__global__ void __launch_bounds__(256)
logsoftmax_q(const float* __restrict__ Qp, __bf16* __restrict__ logpQ) {
  __shared__ float sm[256];
  const size_t row = blockIdx.x;
  const float* x = Qp + row * DMODEL;
  __bf16* y = logpQ + row * DMODEL;
  const int t = threadIdx.x;

  float mx = -3.4e38f;
#pragma unroll
  for (int i = 0; i < DMODEL / 256; ++i) mx = fmaxf(mx, x[t + i * 256]);
  mx = block_max(mx, sm);

  float s = 0.f;
#pragma unroll
  for (int i = 0; i < DMODEL / 256; ++i) s += __expf(x[t + i * 256] - mx);
  s = block_sum(s, sm);
  const float lz = __logf(s);

#pragma unroll
  for (int i = 0; i < DMODEL / 256; ++i)
    y[t + i * 256] = (__bf16)(x[t + i * 256] - mx - lz);
}

// ---------------------------------------------------------------------------
// Kernel 3: pK = softmax(Kp) (bf16), neg_ent[row] = sum pK*logpK (f32).
// ---------------------------------------------------------------------------
__global__ void __launch_bounds__(256)
logsoftmax_k(const float* __restrict__ Kp, __bf16* __restrict__ pK,
             float* __restrict__ neg_ent) {
  __shared__ float sm[256];
  const size_t row = blockIdx.x;
  const float* x = Kp + row * DMODEL;
  __bf16* y = pK + row * DMODEL;
  const int t = threadIdx.x;

  float mx = -3.4e38f;
#pragma unroll
  for (int i = 0; i < DMODEL / 256; ++i) mx = fmaxf(mx, x[t + i * 256]);
  mx = block_max(mx, sm);

  float s = 0.f;
#pragma unroll
  for (int i = 0; i < DMODEL / 256; ++i) s += __expf(x[t + i * 256] - mx);
  s = block_sum(s, sm);
  const float lz = __logf(s);

  float ne = 0.f;
#pragma unroll
  for (int i = 0; i < DMODEL / 256; ++i) {
    const float lp = x[t + i * 256] - mx - lz;
    const float pr = __expf(lp);
    y[t + i * 256] = (__bf16)pr;
    ne += pr * lp;
  }
  ne = block_sum(ne, sm);
  if (t == 0) neg_ent[row] = ne;
}

// ---------------------------------------------------------------------------
// Kernel 4: scores[b,i,j] = sum_d logpQ[b,i,d]*pK[b,j,d] - neg_ent[b,j]
// M=N=2048 per batch, K=1024; A/B already bf16. Output -> d_out attn region.
// ---------------------------------------------------------------------------
__global__ void __launch_bounds__(256)
scores_gemm(const __bf16* __restrict__ logpQ, const __bf16* __restrict__ pK,
            const float* __restrict__ neg_ent, float* __restrict__ attn) {
  __shared__ __align__(16) __bf16 As[2 * BUFE];
  __shared__ __align__(16) __bf16 Bs[2 * BUFE];

  const int b     = blockIdx.z;
  const int tileN = blockIdx.x * 128;
  const int tileM = blockIdx.y * 128;
  const size_t rowOff = (size_t)b * NSEQ;
  const int t    = threadIdx.x;
  const int lane = t & 31;
  const int wv   = t >> 5;
  const int wr   = (wv >> 1) * 32;
  const int wc   = (wv & 1) * 64;

  auto stage = [&](int buf, int kb) {
    unsigned int* A = (unsigned int*)(As + buf * BUFE);
    unsigned int* B = (unsigned int*)(Bs + buf * BUFE);
#pragma unroll
    for (int it = 0; it < 8; ++it) {        // 128 rows x 16 dwords, 8 per thread
      const int idx = t + it * 256;
      const int row = idx >> 4;
      const int c   = idx & 15;
      A[row * (LDSS / 2) + c] =
          ((const unsigned int*)(logpQ + (rowOff + tileM + row) * DMODEL + kb))[c];
      B[row * (LDSS / 2) + c] =
          ((const unsigned int*)(pK + (rowOff + tileN + row) * DMODEL + kb))[c];
    }
  };

  v8f acc[2][4] = {};
  stage(0, 0);
  int p = 0;
  for (int kb = 0; kb < DMODEL; kb += 32) {
    __syncthreads();
    if (kb + 32 < DMODEL) stage(p ^ 1, kb + 32);
    if (kb + 64 < DMODEL) {
      __builtin_prefetch(logpQ + (rowOff + tileM + (t & 127)) * DMODEL + kb + 64, 0, 3);
      __builtin_prefetch(pK    + (rowOff + tileN + (t & 127)) * DMODEL + kb + 64, 0, 3);
    }
    const __bf16* A = As + p * BUFE;
    const __bf16* B = Bs + p * BUFE;
    v16bf af[2], bfrag[4];
    af[0] = load_a_frag(A, wr,      lane);
    af[1] = load_a_frag(A, wr + 16, lane);
#pragma unroll
    for (int n = 0; n < 4; ++n) bfrag[n] = load_b_frag(B, wc + n * 16, lane);
#pragma unroll
    for (int m = 0; m < 2; ++m)
#pragma unroll
      for (int n = 0; n < 4; ++n) acc[m][n] = wmma_bf16(af[m], bfrag[n], acc[m][n]);
    p ^= 1;
  }

  float* out = attn + (size_t)b * NSEQ * NSEQ;
  const int kh   = lane >> 4;
  const int nloc = lane & 15;
#pragma unroll
  for (int m = 0; m < 2; ++m)
#pragma unroll
    for (int n = 0; n < 4; ++n) {
      const int gj = tileN + wc + n * 16 + nloc;
      const float ne = neg_ent[rowOff + gj];
#pragma unroll
      for (int e = 0; e < 8; ++e) {
        const int gi = tileM + wr + m * 16 + kh * 8 + e;
        out[(size_t)gi * NSEQ + gj] = acc[m][n][e] - ne;   // scores = cross - neg_ent
      }
    }
}

// ---------------------------------------------------------------------------
// Kernel 5: in-place softmax over j (rows of length 2048) in d_out attn.
// ---------------------------------------------------------------------------
__global__ void __launch_bounds__(256)
attn_softmax(float* __restrict__ attn) {
  __shared__ float sm[256];
  const size_t row = blockIdx.x;
  float* x = attn + row * NSEQ;
  const int t = threadIdx.x;

  float mx = -3.4e38f;
#pragma unroll
  for (int i = 0; i < NSEQ / 256; ++i) mx = fmaxf(mx, x[t + i * 256]);
  mx = block_max(mx, sm);

  float s = 0.f;
  float e[NSEQ / 256];
#pragma unroll
  for (int i = 0; i < NSEQ / 256; ++i) { e[i] = __expf(x[t + i * 256] - mx); s += e[i]; }
  s = block_sum(s, sm);
  const float inv = 1.0f / s;

#pragma unroll
  for (int i = 0; i < NSEQ / 256; ++i) x[t + i * 256] = e[i] * inv;
}

// ---------------------------------------------------------------------------
// Kernel 6: out[b,i,d] = sum_j attn[b,i,j] * Vp[b,j,d]
// M=2048, N=1024, K=2048. A = attn f32 (convert in staging),
// B = Vp f32 transposed+converted in staging.
// ---------------------------------------------------------------------------
__global__ void __launch_bounds__(256)
out_gemm(const float* __restrict__ attn, const float* __restrict__ Vp,
         float* __restrict__ out) {
  __shared__ __align__(16) __bf16 As[2 * BUFE];
  __shared__ __align__(16) __bf16 Bs[2 * BUFE];

  const int b     = blockIdx.z;
  const int tileN = blockIdx.x * 128;   // d
  const int tileM = blockIdx.y * 128;   // i
  const size_t aOff = (size_t)b * NSEQ * NSEQ;
  const size_t vOff = (size_t)b * NSEQ;
  const int t    = threadIdx.x;
  const int lane = t & 31;
  const int wv   = t >> 5;
  const int wr   = (wv >> 1) * 32;
  const int wc   = (wv & 1) * 64;

  auto stage = [&](int buf, int kb) {
    __bf16* A = As + buf * BUFE;
    __bf16* B = Bs + buf * BUFE;
    // A: attn[tileM+row][kb..kb+31] row-major, f32 -> bf16
#pragma unroll
    for (int it = 0; it < 4; ++it) {
      const int idx = t + it * 256;
      const int row = idx >> 3;
      const int c4  = (idx & 7) * 4;
      float4 va = *(const float4*)(attn + aOff + (size_t)(tileM + row) * NSEQ + kb + c4);
      __bf16* da = A + row * LDSS + c4;
      da[0] = (__bf16)va.x; da[1] = (__bf16)va.y; da[2] = (__bf16)va.z; da[3] = (__bf16)va.w;
    }
    // B: Vp[kb+jl][tileN..tileN+127] -> Bs[n=d_loc][k=j_loc]  (transpose in LDS)
#pragma unroll
    for (int it = 0; it < 4; ++it) {
      const int idx = t + it * 256;       // 32 rows x 32 float4
      const int jl  = idx >> 5;
      const int c4  = (idx & 31) * 4;
      float4 vb = *(const float4*)(Vp + (vOff + kb + jl) * DMODEL + tileN + c4);
      B[(c4 + 0) * LDSS + jl] = (__bf16)vb.x;
      B[(c4 + 1) * LDSS + jl] = (__bf16)vb.y;
      B[(c4 + 2) * LDSS + jl] = (__bf16)vb.z;
      B[(c4 + 3) * LDSS + jl] = (__bf16)vb.w;
    }
  };

  v8f acc[2][4] = {};
  stage(0, 0);
  int p = 0;
  for (int kb = 0; kb < NSEQ; kb += 32) {
    __syncthreads();
    if (kb + 32 < NSEQ) stage(p ^ 1, kb + 32);
    if (kb + 64 < NSEQ) {
      __builtin_prefetch(attn + aOff + (size_t)(tileM + (t & 127)) * NSEQ + kb + 64, 0, 3);
      __builtin_prefetch(Vp + (vOff + kb + 64 + (t & 31)) * DMODEL + tileN, 0, 3);
    }
    const __bf16* A = As + p * BUFE;
    const __bf16* B = Bs + p * BUFE;
    v16bf af[2], bfrag[4];
    af[0] = load_a_frag(A, wr,      lane);
    af[1] = load_a_frag(A, wr + 16, lane);
#pragma unroll
    for (int n = 0; n < 4; ++n) bfrag[n] = load_b_frag(B, wc + n * 16, lane);
#pragma unroll
    for (int m = 0; m < 2; ++m)
#pragma unroll
      for (int n = 0; n < 4; ++n) acc[m][n] = wmma_bf16(af[m], bfrag[n], acc[m][n]);
    p ^= 1;
  }

  float* o = out + (size_t)b * NSEQ * DMODEL;
  const int kh   = lane >> 4;
  const int nloc = lane & 15;
#pragma unroll
  for (int m = 0; m < 2; ++m)
#pragma unroll
    for (int n = 0; n < 4; ++n) {
      const int gd = tileN + wc + n * 16 + nloc;
#pragma unroll
      for (int e = 0; e < 8; ++e) {
        const int gi = tileM + wr + m * 16 + kh * 8 + e;
        o[(size_t)gi * DMODEL + gd] = acc[m][n][e];
      }
    }
}

// ---------------------------------------------------------------------------
// Host-side launcher
// ---------------------------------------------------------------------------
extern "C" void kernel_launch(void* const* d_in, const int* in_sizes, int n_in,
                              void* d_out, int out_size, void* d_ws, size_t ws_size,
                              hipStream_t stream) {
  const float* Q  = (const float*)d_in[0];
  const float* K  = (const float*)d_in[1];
  const float* V  = (const float*)d_in[2];
  const float* Wq = (const float*)d_in[3];
  const float* bq = (const float*)d_in[4];
  const float* Wk = (const float*)d_in[5];
  const float* bk = (const float*)d_in[6];
  const float* Wv = (const float*)d_in[7];
  const float* bv = (const float*)d_in[8];

  const size_t ROWS = (size_t)BATCH * NSEQ;              // 8192
  float* outP  = (float*)d_out;                          // [B,NQ,D]
  float* attnP = outP + ROWS * DMODEL;                   // [B,NQ,NK]

  char* ws = (char*)d_ws;
  size_t off = 0;
  auto alloc = [&](size_t bytes) { char* p = ws + off; off = (off + bytes + 255) & ~(size_t)255; return p; };
  float*  Qp      = (float*)alloc(ROWS * DMODEL * sizeof(float));
  float*  Kp      = (float*)alloc(ROWS * DMODEL * sizeof(float));
  float*  Vp      = (float*)alloc(ROWS * DMODEL * sizeof(float));
  __bf16* logpQ   = (__bf16*)alloc(ROWS * DMODEL * sizeof(__bf16));
  __bf16* pK      = (__bf16*)alloc(ROWS * DMODEL * sizeof(__bf16));
  float*  neg_ent = (float*)alloc(ROWS * sizeof(float));
  (void)ws_size; (void)in_sizes; (void)n_in; (void)out_size;

  // 1) projections  (M=8192, N=1024, K=1024), 128x128 tiles
  dim3 pg(DMODEL / 128, ROWS / 128);
  proj_gemm<<<pg, 256, 0, stream>>>(Q, Wq, bq, Qp);
  proj_gemm<<<pg, 256, 0, stream>>>(K, Wk, bk, Kp);
  proj_gemm<<<pg, 256, 0, stream>>>(V, Wv, bv, Vp);

  // 2,3) row log-softmaxes
  logsoftmax_q<<<dim3(ROWS), 256, 0, stream>>>(Qp, logpQ);
  logsoftmax_k<<<dim3(ROWS), 256, 0, stream>>>(Kp, pK, neg_ent);

  // 4) scores -> attn region of d_out  (2048x2048x1024 per batch)
  scores_gemm<<<dim3(NSEQ / 128, NSEQ / 128, BATCH), 256, 0, stream>>>(logpQ, pK, neg_ent, attnP);

  // 5) softmax over j, in place
  attn_softmax<<<dim3(ROWS), 256, 0, stream>>>(attnP);

  // 6) out = attn @ Vp  (2048x1024x2048 per batch)
  out_gemm<<<dim3(DMODEL / 128, NSEQ / 128, BATCH), 256, 0, stream>>>(attnP, Vp, outP);
}
